// MaskedMultiHeadAttentionBlock_2052994367506
// MI455X (gfx1250) — matrix operand order repaired and checked by reading, not compiled, not run on hardware
//
#include <hip/hip_runtime.h>
#include <hip/hip_bf16.h>
#include <stdint.h>

// MaskedMultiHeadAttentionBlock for MI455X (gfx1250, wave32, WMMA bf16).
// Pipeline:
//   1) gemm_bias<fp32->bf16> : Q = x@Wq^T+b   [8192x512]  (bf16 ws)
//   2) gemm_bias<fp32->bf16> : K = x@Wk^T+b   [8192x512]  (bf16 ws)
//   3) gemm_bias<fp32->bf16> : V = x@Wv^T+b   [8192x1024] (bf16 ws)
//   4) attn_stats : per (b,h) per key-column online max / sum-exp over q axis
//      (softmax in the reference is over the QUERY axis => column softmax)
//   5) attn_out   : recompute scores, column-normalize, P@V via WMMA -> bf16 ws
//   6) gemm_bias<bf16->fp32> : out = att@Wo^T+b -> d_out
// The flat .view() reshapes mean head (b,h) of Q is the contiguous block at
// Qws + b*C*512 + h*C*64 (row-major [C x 64]); same pattern for K/V/attn-out.
//
// GEMM uses 64x128 block tiles: each wave holds one W B-fragment and reuses it
// across 4 row-subtiles (4 WMMAs per 16-dword B load) for arithmetic intensity.

#define DEVINL __device__ __forceinline__

typedef __attribute__((ext_vector_type(16))) __bf16 v16bf;
typedef __attribute__((ext_vector_type(8)))  float  v8f;

namespace {

constexpr int Bb  = 4;
constexpr int Cc  = 2048;
constexpr int Hh  = 8;
constexpr int DK  = 64;
constexpr int DV  = 128;
constexpr int DIN = 1024;
constexpr int Nrows = Bb * Cc;          // 8192
constexpr float SCALE = 0.125f;         // 1/sqrt(DK)

union BF16Frag { uint32_t u[8]; v16bf v; };

DEVINL uint16_t f2bf(float f) {
  union { float f; uint32_t u; } c; c.f = f;
  uint32_t u = c.u;
  return (uint16_t)((u + 0x7FFFu + ((u >> 16) & 1u)) >> 16);  // RNE-ish
}
DEVINL uint32_t pack2(float lo, float hi) {
  return (uint32_t)f2bf(lo) | ((uint32_t)f2bf(hi) << 16);
}
DEVINL int kmapPair(int p, int hs) {
  // Even K index of bf16 pair p (0..7) for lane half hs; CDNA5 16-bit
  // A-matrix 16x32 VGPR layout (CDNA5 ISA sec 7.12.2).
  return ((p & 4) ? 16 : 0) + (hs ? 8 : 0) + 2 * (p & 3);
}

DEVINL v8f wmma_bf16(v16bf a, v16bf b, v8f c) {
  return __builtin_amdgcn_wmma_f32_16x16x32_bf16(
      /*neg_a=*/false, a, /*neg_b=*/false, b,
      /*c_mod=*/(short)0, c, /*reuse_a=*/false, /*reuse_b=*/false);
}

// A-fragment (16x32 MxK) from row-major bf16 buffer, ld elements per row.
// Lane: M = lane&15, K pairs chosen by hs = lane>>4.
DEVINL v16bf load_afrag(const uint16_t* base, int row, int ld, int k0, int hs) {
  BF16Frag r;
#pragma unroll
  for (int p = 0; p < 8; ++p)
    r.u[p] = *reinterpret_cast<const uint32_t*>(base + row * ld + k0 + kmapPair(p, hs));
  return r.v;
}

// B-fragment (32x16 KxN), lane = N column (lane&15), elements = K pairs via kmapPair(p, hs).
// Case 1: B[k][n] = W[o][k0+k] with o = oBase+n, W fp32 row-major [O x ldk].
DEVINL v16bf load_bfrag_w(const float* W, int o, int ldk, int k0, int hs) {
  BF16Frag r;
#pragma unroll
  for (int p = 0; p < 8; ++p)
    r.u[p] = pack2(W[o * ldk + k0 + kmapPair(p, hs)], W[o * ldk + k0 + kmapPair(p, hs) + 1]);
  return r.v;
}
// Case 2: B = K^T: B[d][n] = Kh[kColBase+n][d], Kh bf16 row-major [C x 64].
DEVINL v16bf load_bfrag_kT(const uint16_t* Kh, int kColBase, int d) {
  BF16Frag r;
#pragma unroll
  for (int p = 0; p < 8; ++p) {
    uint32_t lo = Kh[(kColBase + 2 * p) * DK + d];
    uint32_t hi = Kh[(kColBase + 2 * p + 1) * DK + d];
    r.u[p] = lo | (hi << 16);
  }
  return r.v;
}
// Case 3: V tile: lane = d column, elements = k rows (stride DV) via kmapPair(p, hs).
DEVINL v16bf load_bfrag_row(const uint16_t* colptr, int hs) {
  BF16Frag r;
#pragma unroll
  for (int p = 0; p < 8; ++p) {
    uint32_t lo = colptr[(size_t)kmapPair(p, hs) * DV];
    uint32_t hi = colptr[(size_t)(kmapPair(p, hs) + 1) * DV];
    r.u[p] = lo | (hi << 16);
  }
  return r.v;
}

// ---------------------------------------------------------------------------
// Generic  out[N x O] = A[N x K] @ W[O x K]^T + bias  via WMMA bf16.
// Block: 256 threads (8 waves). Block tile: 64 rows x 128 cols; wave w owns
// 16 columns and 4 row-subtiles -> one B-fragment feeds 4 WMMAs per K step.
// Shared A tile (64x32 bf16, 4KB) staged through LDS with b128 transfers.
// ---------------------------------------------------------------------------
template <bool A_IS_BF16, bool OUT_BF16>
__global__ void gemm_bias_kernel(const void* __restrict__ A_,
                                 const float* __restrict__ W,
                                 const float* __restrict__ bias,
                                 void* __restrict__ out_,
                                 int K, int O) {
  __shared__ uint16_t Atile[64 * 32];
  const int tid  = threadIdx.x;
  const int wave = tid >> 5;
  const int lane = tid & 31;
  const int hs   = lane >> 4;
  const int nidx = lane & 15;
  const int rowBase = blockIdx.x * 64;
  const int colBase = blockIdx.y * 128 + wave * 16;

  v8f acc[4] = {{}, {}, {}, {}};
  for (int k0 = 0; k0 < K; k0 += 32) {
    {  // stage A tile (64x32) as bf16 in LDS; 8 consecutive elements/thread
      const int e0 = tid * 8;          // 2048 elements total
      const int r  = e0 >> 5;
      const int kk = e0 & 31;
      if (A_IS_BF16) {
        const uint16_t* A = (const uint16_t*)A_;
        *reinterpret_cast<uint4*>(&Atile[e0]) =
            *reinterpret_cast<const uint4*>(&A[(size_t)(rowBase + r) * K + k0 + kk]);
      } else {
        const float* A  = (const float*)A_;
        const float4 lo = *reinterpret_cast<const float4*>(
            &A[(size_t)(rowBase + r) * K + k0 + kk]);
        const float4 hi = *reinterpret_cast<const float4*>(
            &A[(size_t)(rowBase + r) * K + k0 + kk + 4]);
        uint4 pk;
        pk.x = pack2(lo.x, lo.y);
        pk.y = pack2(lo.z, lo.w);
        pk.z = pack2(hi.x, hi.y);
        pk.w = pack2(hi.z, hi.w);
        *reinterpret_cast<uint4*>(&Atile[e0]) = pk;
      }
    }
    __syncthreads();
    const v16bf b = load_bfrag_w(W, colBase + nidx, K, k0, hs);
#pragma unroll
    for (int t = 0; t < 4; ++t) {
      v16bf a = load_afrag(Atile, t * 16 + nidx, 32, 0, hs);
      acc[t] = wmma_bf16(a, b, acc[t]);
    }
    __syncthreads();
  }

  const float bv  = bias[colBase + nidx];
  const int   col = colBase + nidx;
#pragma unroll
  for (int t = 0; t < 4; ++t) {
#pragma unroll
    for (int r = 0; r < 8; ++r) {
      const int   row = rowBase + t * 16 + r + 8 * hs;
      const float v   = acc[t][r] + bv;
      if (OUT_BF16) ((uint16_t*)out_)[(size_t)row * O + col] = f2bf(v);
      else          ((float*)out_)[(size_t)row * O + col]    = v;
    }
  }
}

// ---------------------------------------------------------------------------
// Pass A: per (b,h), per key-column online (max, sum-exp) over the QUERY axis
// of t[q,k] = (q.k + mask[b,q]) * SCALE.  Grid: (C/256, B*H); each wave owns
// 32 key columns (two 16-col subtiles) so the loop-invariant K^T B-fragments
// amortize each q-tile's Q A-fragment loads over 4 WMMAs.
// ---------------------------------------------------------------------------
__global__ void attn_stats_kernel(const uint16_t* __restrict__ Qws,
                                  const uint16_t* __restrict__ Kws,
                                  const float* __restrict__ mask,
                                  float* __restrict__ stats_m,
                                  float* __restrict__ stats_d) {
  const int tid  = threadIdx.x;
  const int wave = tid >> 5;
  const int lane = tid & 31;
  const int hs   = lane >> 4;
  const int nidx = lane & 15;
  const int bh = blockIdx.y;
  const int b  = bh >> 3;
  const int h  = bh & 7;
  const int kBase = blockIdx.x * 256 + wave * 32;

  const uint16_t* Qh = Qws + (size_t)b * Cc * 512 + (size_t)h * Cc * DK;
  const uint16_t* Kh = Kws + (size_t)b * Cc * 512 + (size_t)h * Cc * DK;

  // Loop-invariant B = K^T fragments: 2 column subtiles x 2 d-halves.
  v16bf bk[2][2];
#pragma unroll
  for (int ct = 0; ct < 2; ++ct) {
    bk[ct][0] = load_afrag(Kh, kBase + ct * 16 + nidx, DK, 0, hs);   // d = 0..31
    bk[ct][1] = load_afrag(Kh, kBase + ct * 16 + nidx, DK, 32, hs);  // d = 32..63
  }

  float run_m[2] = {-1e30f, -1e30f};
  float run_d[2] = {0.0f, 0.0f};
  for (int q0 = 0; q0 < Cc; q0 += 16) {
    const v16bf a0 = load_afrag(Qh, q0 + nidx, DK, 0, hs);
    const v16bf a1 = load_afrag(Qh, q0 + nidx, DK, 32, hs);
    float mv[8];
#pragma unroll
    for (int r = 0; r < 8; ++r) mv[r] = mask[b * Cc + q0 + r + 8 * hs];

#pragma unroll
    for (int ct = 0; ct < 2; ++ct) {
      v8f s = {};
      s = wmma_bf16(a0, bk[ct][0], s);
      s = wmma_bf16(a1, bk[ct][1], s);

      float tv[8];
      float lmax = -1e30f;
#pragma unroll
      for (int r = 0; r < 8; ++r) {
        const float t = (s[r] + mv[r]) * SCALE;
        tv[r] = t;
        lmax  = fmaxf(lmax, t);
      }
      // combine the two lane halves holding the same column (lane ^ 16)
      const float tmax = fmaxf(lmax, __shfl_xor(lmax, 16, 32));
      const float nm   = fmaxf(run_m[ct], tmax);
      float ls = 0.0f;
#pragma unroll
      for (int r = 0; r < 8; ++r) ls += __expf(tv[r] - nm);
      ls += __shfl_xor(ls, 16, 32);
      run_d[ct] = run_d[ct] * __expf(run_m[ct] - nm) + ls;
      run_m[ct] = nm;
    }
  }
  if (hs == 0) {
#pragma unroll
    for (int ct = 0; ct < 2; ++ct) {
      stats_m[bh * Cc + kBase + ct * 16 + nidx] = run_m[ct];
      stats_d[bh * Cc + kBase + ct * 16 + nidx] = run_d[ct];
    }
  }
}

// ---------------------------------------------------------------------------
// Pass B: out[q,d] = sum_k exp(t[q,k]-m[k])/d[k] * V[k,d].
// Grid: (C/16, B*H). Block: 8 waves; wave w owns 16 of the 128 d columns and
// computes 32 of each 256-wide key chunk's P columns; P transits LDS to be
// re-read in A-fragment layout for the P@V WMMA (shared by all 8 waves).
// ---------------------------------------------------------------------------
__global__ void attn_out_kernel(const uint16_t* __restrict__ Qws,
                                const uint16_t* __restrict__ Kws,
                                const uint16_t* __restrict__ Vws,
                                const float* __restrict__ mask,
                                const float* __restrict__ stats_m,
                                const float* __restrict__ stats_d,
                                uint16_t* __restrict__ AttOut) {
  __shared__ uint16_t Pbuf[16 * 256];
  const int tid  = threadIdx.x;
  const int wave = tid >> 5;
  const int lane = tid & 31;
  const int hs   = lane >> 4;
  const int nidx = lane & 15;
  const int bh = blockIdx.y;
  const int b  = bh >> 3;
  const int h  = bh & 7;
  const int q0 = blockIdx.x * 16;

  const uint16_t* Qh = Qws + (size_t)b * Cc * 512 + (size_t)h * Cc * DK;
  const uint16_t* Kh = Kws + (size_t)b * Cc * 512 + (size_t)h * Cc * DK;
  const uint16_t* Vh = Vws + (size_t)b * Cc * 1024 + (size_t)h * Cc * DV;

  const v16bf aq0 = load_afrag(Qh, q0 + nidx, DK, 0, hs);
  const v16bf aq1 = load_afrag(Qh, q0 + nidx, DK, 32, hs);
  float mrow[8];
#pragma unroll
  for (int r = 0; r < 8; ++r)
    mrow[r] = mask[b * Cc + q0 + r + 8 * hs] * SCALE;

  v8f accO = {};
  for (int kb = 0; kb < Cc; kb += 256) {
    if (kb + 256 < Cc)  // speculative prefetch of the next V chunk's rows
      __builtin_prefetch(Vh + (size_t)(kb + 256 + lane) * DV + wave * 16, 0, 1);

    // ---- scores + column-normalized P for this wave's 32 key columns ----
#pragma unroll
    for (int khalf = 0; khalf < 2; ++khalf) {
      const int kcb = kb + wave * 32 + khalf * 16;
      v16bf bk0 = load_afrag(Kh, kcb + nidx, DK, 0, hs);
      v16bf bk1 = load_afrag(Kh, kcb + nidx, DK, 32, hs);
      v8f s = {};
      s = wmma_bf16(aq0, bk0, s);
      s = wmma_bf16(aq1, bk1, s);
      const int   kcol = kcb + nidx;
      const float mk   = stats_m[bh * Cc + kcol];
      const float invd = 1.0f / stats_d[bh * Cc + kcol];
#pragma unroll
      for (int r = 0; r < 8; ++r) {
        const float p = __expf(s[r] * SCALE + mrow[r] - mk) * invd;
        Pbuf[(r + 8 * hs) * 256 + wave * 32 + khalf * 16 + nidx] = f2bf(p);
      }
    }
    __syncthreads();

    // ---- accO += P[16 x 256] @ V[256 x 16] (this wave's d columns) ----
#pragma unroll
    for (int kk = 0; kk < 8; ++kk) {
      v16bf ap = load_afrag(Pbuf, nidx, 256, kk * 32, hs);
      const uint16_t* vptr = Vh + (size_t)(kb + kk * 32) * DV + wave * 16 + nidx;
      v16bf bv = load_bfrag_row(vptr, hs);
      accO = wmma_bf16(ap, bv, accO);
    }
    __syncthreads();
  }

  const int dcol = wave * 16 + nidx;
#pragma unroll
  for (int r = 0; r < 8; ++r) {
    const int q = q0 + r + 8 * hs;
    AttOut[(size_t)b * Cc * 1024 + (size_t)h * Cc * DV + (size_t)q * DV + dcol] =
        f2bf(accO[r]);
  }
}

}  // namespace

extern "C" void kernel_launch(void* const* d_in, const int* in_sizes, int n_in,
                              void* d_out, int out_size, void* d_ws, size_t ws_size,
                              hipStream_t stream) {
  (void)in_sizes; (void)n_in; (void)out_size; (void)ws_size;
  const float* x    = (const float*)d_in[0];
  const float* mask = (const float*)d_in[1];
  const float* Mq_w = (const float*)d_in[2];
  const float* Mq_b = (const float*)d_in[3];
  const float* Mk_w = (const float*)d_in[4];
  const float* Mk_b = (const float*)d_in[5];
  const float* Mv_w = (const float*)d_in[6];
  const float* Mv_b = (const float*)d_in[7];
  const float* Wo_w = (const float*)d_in[8];
  const float* Wo_b = (const float*)d_in[9];
  float* out = (float*)d_out;

  // workspace carve-up (bf16 activations + fp32 column stats, ~48.5 MB)
  char* w = (char*)d_ws;
  uint16_t* Qws    = (uint16_t*)w;                    w += (size_t)Nrows * 512 * 2;
  uint16_t* Kws    = (uint16_t*)w;                    w += (size_t)Nrows * 512 * 2;
  uint16_t* Vws    = (uint16_t*)w;                    w += (size_t)Nrows * 1024 * 2;
  uint16_t* AttOut = (uint16_t*)w;                    w += (size_t)Nrows * 1024 * 2;
  float*    st_m   = (float*)w;                       w += (size_t)Bb * Hh * Cc * 4;
  float*    st_d   = (float*)w;

  const dim3 blk(256);

  // 1-3) Q / K / V projections (fp32 in, bf16 out)
  gemm_bias_kernel<false, true><<<dim3(Nrows / 64, 512 / 128), blk, 0, stream>>>(
      (const void*)x, Mq_w, Mq_b, (void*)Qws, DIN, 512);
  gemm_bias_kernel<false, true><<<dim3(Nrows / 64, 512 / 128), blk, 0, stream>>>(
      (const void*)x, Mk_w, Mk_b, (void*)Kws, DIN, 512);
  gemm_bias_kernel<false, true><<<dim3(Nrows / 64, 1024 / 128), blk, 0, stream>>>(
      (const void*)x, Mv_w, Mv_b, (void*)Vws, DIN, 1024);

  // 4) per-key-column softmax stats over the query axis
  attn_stats_kernel<<<dim3(Cc / 256, Bb * Hh), blk, 0, stream>>>(
      Qws, Kws, mask, st_m, st_d);

  // 5) attention output (column-normalized P @ V)
  attn_out_kernel<<<dim3(Cc / 16, Bb * Hh), blk, 0, stream>>>(
      Qws, Kws, Vws, mask, st_m, st_d, AttOut);

  // 6) output projection (bf16 in, fp32 out)
  gemm_bias_kernel<true, false><<<dim3(Nrows / 64, 1024 / 128), blk, 0, stream>>>(
      (const void*)AttOut, Wo_w, Wo_b, (void*)out, 1024, 1024);
}
